// LoCS_87454124081352
// MI455X (gfx1250) — matrix operand order repaired and checked by reading, not compiled, not run on hardware
//
#include <hip/hip_runtime.h>
#include <cstdint>
#include <cstddef>

#define DEVINL __device__ __forceinline__

// ---------------------------------------------------------------- types
typedef __attribute__((ext_vector_type(16))) __bf16 bf16x16;
typedef __attribute__((ext_vector_type(8)))  float  f32x8;
typedef __attribute__((ext_vector_type(4)))  unsigned int u32x4;
typedef __attribute__((ext_vector_type(8)))  int    i32x8;
typedef __attribute__((ext_vector_type(4)))  int    i32x4;

union FragBF {
  bf16x16 v;
  uint32_t u[8];
};

// ---------------------------------------------------------------- helpers
DEVINL uint16_t f2bf_bits(float f) {
  union { float f; uint32_t u; } x; x.f = f;
  uint32_t r = x.u + 0x7FFFu + ((x.u >> 16) & 1u);  // round-to-nearest-even
  return (uint16_t)(r >> 16);
}
DEVINL float bf_bits2f(uint16_t h) {
  union { uint32_t u; float f; } y; y.u = ((uint32_t)h) << 16;
  return y.f;
}
DEVINL float elu_f(float x)  { return x > 0.f ? x : __expf(x) - 1.f; }
DEVINL float sigm_f(float x) { return 1.f / (1.f + __expf(-x)); }

// A fragment: 16x32 bf16 slice. Lane holds row m=lane&15; K pattern per ISA
// dense-A table (VGPR0-3: K=0..7 / 8..15 by lane half; VGPR4-7: +16).
DEVINL FragBF load_a(const uint32_t* p, int row, int stride_bf, int k0, int lane) {
  FragBF f;
  const int base = row * stride_bf + k0;
  const int kb = (lane & 16) ? 8 : 0;
#pragma unroll
  for (int v = 0; v < 8; ++v) {
    const int k = ((v & 4) ? 16 : 0) + kb + (v & 3) * 2;
    f.u[v] = p[(base + k) >> 1];
  }
  return f;
}
// B fragment: 32x16. Lane holds column n=lane&15 (= row n of row-major W[N][K]);
// K pattern matches A: VGPR0-3: K=0..7 / 8..15 by lane half; VGPR4-7: +16.
DEVINL FragBF load_b(const uint32_t* p, int nrow, int stride_bf, int k0, int lane) {
  FragBF f;
  const int base = nrow * stride_bf + k0;
  const int kb = (lane & 16) ? 8 : 0;
#pragma unroll
  for (int v = 0; v < 8; ++v) {
    const int k = ((v & 4) ? 16 : 0) + kb + (v & 3) * 2;
    f.u[v] = p[(base + k) >> 1];
  }
  return f;
}
DEVINL f32x8 wmma_bf16(FragBF a, FragBF b, f32x8 c) {
  return __builtin_amdgcn_wmma_f32_16x16x32_bf16(false, a.v, false, b.v,
                                                 (short)0, c, false, false);
}

// ---------------------------------------------------------------- problem dims
static constexpr int BB = 8, VV = 16, TT = 64;
static constexpr int DIN = 6, HH = 256, RH = 64, KK = 2, TMP = 256;
static constexpr int EE = VV * (VV - 1);          // 240
static constexpr int NREL = 7, NPOS = 3;
static constexpr int EROWS = BB * EE * TT;        // 122880 edge-time rows
static constexpr int NROWS = BB * VV * TT;        // 8192  node-time rows
static constexpr int NSEQ  = BB * EE;             // 1920  LSTM sequences
static constexpr int OUT_HALF = BB * TT * EE * KK; // 245760

// ---------------------------------------------------------------- ws layout (bf16 elems)
static constexpr size_t O_EFW1 = 0;                       // 256x32 (padded from 256x16)
static constexpr size_t O_EFW2 = O_EFW1 + 256 * 32;
static constexpr size_t O_M3W1 = O_EFW2 + 256 * 256;
static constexpr size_t O_M3W2 = O_M3W1 + 256 * 256;
static constexpr size_t O_M4W1 = O_M3W2 + 256 * 256;      // 256x768
static constexpr size_t O_M4W2 = O_M4W1 + 256 * 768;
static constexpr size_t O_FWIH = O_M4W2 + 256 * 256;
static constexpr size_t O_RWIH = O_FWIH + 256 * 256;
static constexpr size_t O_FWHH = O_RWIH + 256 * 256;      // 256x64
static constexpr size_t O_RWHH = O_FWHH + 256 * 64;
static constexpr size_t O_EW1  = O_RWHH + 256 * 64;       // 256x128
static constexpr size_t O_EDGE = O_EW1  + 256 * 128;      // edge_attr   [EROWS][256]
static constexpr size_t O_NIN  = O_EDGE + (size_t)EROWS * 256;  // node in  [NROWS][256]
static constexpr size_t O_NODE = O_NIN  + (size_t)NROWS * 256;  // node out [NROWS][256]
static constexpr size_t O_XE   = O_NODE + (size_t)NROWS * 256;  // mlp4 out [EROWS][256]
static constexpr size_t O_XWF  = O_XE   + (size_t)EROWS * 256;  // [EROWS][256]
static constexpr size_t O_XWR  = O_XWF  + (size_t)EROWS * 256;
static constexpr size_t O_FWD  = O_XWR  + (size_t)EROWS * 256;  // [EROWS][64]
static constexpr size_t O_REV  = O_FWD  + (size_t)EROWS * 64;

// ---------------------------------------------------------------- weight conversion
__global__ __launch_bounds__(256) void k_f2bf(const float* __restrict__ s,
                                              uint16_t* __restrict__ d, int n) {
  int i = blockIdx.x * 256 + threadIdx.x;
  if (i < n) d[i] = f2bf_bits(s[i]);
}
__global__ __launch_bounds__(256) void k_pad_efw1(const float* __restrict__ s,
                                                  uint16_t* __restrict__ d) {
  int i = blockIdx.x * 256 + threadIdx.x;   // 256*32
  if (i < 256 * 32) {
    int n = i >> 5, k = i & 31;
    d[i] = (k < 16) ? f2bf_bits(s[n * 16 + k]) : (uint16_t)0;
  }
}

// ---------------------------------------------------------------- S1: edge filter MLP (16->256->256)
__global__ __launch_bounds__(256) void k_edge_mlp(
    const float* __restrict__ raw, const float* __restrict__ pos,
    const uint32_t* __restrict__ w1, const float* __restrict__ b1,
    const uint32_t* __restrict__ w2, const float* __restrict__ b2,
    uint16_t* __restrict__ outbf) {
  __shared__ __bf16 sA[16 * 32];
  __shared__ __bf16 sH[16 * 256];
  const int row0 = blockIdx.x * 16;
  const int tid = threadIdx.x, lane = tid & 31, wave = tid >> 5;

  for (int i = tid; i < 16 * 32; i += 256) {
    const int m = i >> 5, k = i & 31;
    const int row = row0 + m;
    float val = 0.f;
    if (k < NREL + DIN)      val = raw[(size_t)row * (NREL + DIN) + k];
    else if (k < 16)         val = pos[(size_t)row * NPOS + (k - (NREL + DIN))];
    union { uint16_t u; __bf16 b; } c; c.u = f2bf_bits(val);
    sA[i] = c.b;
  }
  __syncthreads();

  const uint32_t* sA32 = (const uint32_t*)sA;
  const uint32_t* sH32 = (const uint32_t*)sH;
#pragma unroll
  for (int i = 0; i < 2; ++i) {                    // layer 1, K=32 (zero padded)
    const int n0 = (wave * 2 + i) * 16;
    f32x8 acc = {};
    FragBF a = load_a(sA32, lane & 15, 32, 0, lane);
    FragBF b = load_b(w1, n0 + (lane & 15), 32, 0, lane);
    acc = wmma_bf16(a, b, acc);
    const int n = n0 + (lane & 15);
#pragma unroll
    for (int r = 0; r < 8; ++r) {
      const int m = r + ((lane & 16) ? 8 : 0);
      const float z = elu_f(acc[r] + b1[n]);
      union { uint16_t u; __bf16 b; } c; c.u = f2bf_bits(z);
      sH[m * 256 + n] = c.b;
    }
  }
  __syncthreads();
#pragma unroll
  for (int i = 0; i < 2; ++i) {                    // layer 2, K=256
    const int n0 = (wave * 2 + i) * 16;
    f32x8 acc = {};
#pragma unroll
    for (int kk = 0; kk < 256; kk += 32) {
      FragBF a = load_a(sH32, lane & 15, 256, kk, lane);
      FragBF b = load_b(w2, n0 + (lane & 15), 256, kk, lane);
      acc = wmma_bf16(a, b, acc);
    }
    const int n = n0 + (lane & 15);
#pragma unroll
    for (int r = 0; r < 8; ++r) {
      const int m = r + ((lane & 16) ? 8 : 0);
      const float z = elu_f(acc[r] + b2[n]);
      outbf[(size_t)(row0 + m) * 256 + n] = f2bf_bits(z);
    }
  }
}

// ---------------------------------------------------------------- S2: edge->node mean + residual
__global__ __launch_bounds__(256) void k_aggregate(
    const uint16_t* __restrict__ edgebf, const float* __restrict__ rel_feat,
    const float* __restrict__ res_w, const float* __restrict__ res_b,
    uint16_t* __restrict__ nodein) {
  const int nrow = blockIdx.x;               // (b*V+v)*T + t
  const int h = threadIdx.x;
  const int t = nrow & (TT - 1);
  const int bv = nrow >> 6;
  const int v = bv & (VV - 1);
  const int b = bv >> 4;
  float s = 0.f;
#pragma unroll
  for (int snd = 0; snd < VV; ++snd) {
    if (snd == v) continue;
    const int e = snd * (VV - 1) + (v < snd ? v : v - 1);
    const size_t erow = ((size_t)(b * EE + e) * TT + t);
    s += bf_bits2f(edgebf[erow * 256 + h]);
  }
  float x = s / (float)(VV - 1) + res_b[h];
#pragma unroll
  for (int d = 0; d < DIN; ++d)
    x += rel_feat[(size_t)nrow * DIN + d] * res_w[h * DIN + d];
  nodein[(size_t)nrow * 256 + h] = f2bf_bits(x);
}

// ---------------------------------------------------------------- S3: node MLP (256->256->256)
__global__ __launch_bounds__(256) void k_mlp2_256(
    const uint32_t* __restrict__ Abf,
    const uint32_t* __restrict__ w1, const float* __restrict__ b1,
    const uint32_t* __restrict__ w2, const float* __restrict__ b2,
    uint16_t* __restrict__ outbf) {
  __shared__ __bf16 sH[16 * 256];
  const int row0 = blockIdx.x * 16;
  const int tid = threadIdx.x, lane = tid & 31, wave = tid >> 5;
  const uint32_t* sH32 = (const uint32_t*)sH;
#pragma unroll
  for (int i = 0; i < 2; ++i) {
    const int n0 = (wave * 2 + i) * 16;
    f32x8 acc = {};
#pragma unroll
    for (int kk = 0; kk < 256; kk += 32) {
      FragBF a = load_a(Abf, row0 + (lane & 15), 256, kk, lane);
      FragBF b = load_b(w1, n0 + (lane & 15), 256, kk, lane);
      acc = wmma_bf16(a, b, acc);
    }
    const int n = n0 + (lane & 15);
#pragma unroll
    for (int r = 0; r < 8; ++r) {
      const int m = r + ((lane & 16) ? 8 : 0);
      const float z = elu_f(acc[r] + b1[n]);
      union { uint16_t u; __bf16 b; } c; c.u = f2bf_bits(z);
      sH[m * 256 + n] = c.b;
    }
  }
  __syncthreads();
#pragma unroll
  for (int i = 0; i < 2; ++i) {
    const int n0 = (wave * 2 + i) * 16;
    f32x8 acc = {};
#pragma unroll
    for (int kk = 0; kk < 256; kk += 32) {
      FragBF a = load_a(sH32, lane & 15, 256, kk, lane);
      FragBF b = load_b(w2, n0 + (lane & 15), 256, kk, lane);
      acc = wmma_bf16(a, b, acc);
    }
    const int n = n0 + (lane & 15);
#pragma unroll
    for (int r = 0; r < 8; ++r) {
      const int m = r + ((lane & 16) ? 8 : 0);
      const float z = elu_f(acc[r] + b2[n]);
      outbf[(size_t)(row0 + m) * 256 + n] = f2bf_bits(z);
    }
  }
}

// ---------------------------------------------------------------- S4: gather-concat MLP (768->256->256)
__global__ __launch_bounds__(256) void k_mlp4(
    const uint32_t* __restrict__ nodebf, const uint32_t* __restrict__ edgebf,
    const uint32_t* __restrict__ w1, const float* __restrict__ b1,
    const uint32_t* __restrict__ w2, const float* __restrict__ b2,
    uint16_t* __restrict__ outbf) {
  __shared__ __bf16 sA[16 * 768];
  __shared__ __bf16 sH[16 * 256];
  __shared__ int sSend[16], sRecv[16];
  const int row0 = blockIdx.x * 16;
  const int tid = threadIdx.x, lane = tid & 31, wave = tid >> 5;

  if (tid < 16) {
    const int row = row0 + tid;
    const int t = row & (TT - 1);
    const int be = row >> 6;
    const int e = be % EE;
    const int b = be / EE;
    const int s = e / (VV - 1);
    const int j = e % (VV - 1);
    const int r = (j < s) ? j : j + 1;
    sSend[tid] = (b * VV + s) * TT + t;
    sRecv[tid] = (b * VV + r) * TT + t;
  }
  __syncthreads();

  uint32_t* sA32w = (uint32_t*)sA;
  for (int i = tid; i < 16 * 384; i += 256) {      // 16 rows x 384 dwords (768 bf16)
    const int m = i / 384, kd = i % 384;
    uint32_t val;
    if (kd < 128)       val = nodebf[(size_t)sSend[m] * 128 + kd];
    else if (kd < 256)  val = nodebf[(size_t)sRecv[m] * 128 + (kd - 128)];
    else                val = edgebf[(size_t)(row0 + m) * 128 + (kd - 256)];
    sA32w[m * 384 + kd] = val;
  }
  __syncthreads();

  const uint32_t* sA32 = (const uint32_t*)sA;
  const uint32_t* sH32 = (const uint32_t*)sH;
#pragma unroll
  for (int i = 0; i < 2; ++i) {                    // layer 1, K=768
    const int n0 = (wave * 2 + i) * 16;
    f32x8 acc = {};
#pragma unroll 4
    for (int kk = 0; kk < 768; kk += 32) {
      FragBF a = load_a(sA32, lane & 15, 768, kk, lane);
      FragBF b = load_b(w1, n0 + (lane & 15), 768, kk, lane);
      acc = wmma_bf16(a, b, acc);
    }
    const int n = n0 + (lane & 15);
#pragma unroll
    for (int r = 0; r < 8; ++r) {
      const int m = r + ((lane & 16) ? 8 : 0);
      const float z = elu_f(acc[r] + b1[n]);
      union { uint16_t u; __bf16 b; } c; c.u = f2bf_bits(z);
      sH[m * 256 + n] = c.b;
    }
  }
  __syncthreads();
#pragma unroll
  for (int i = 0; i < 2; ++i) {                    // layer 2, K=256
    const int n0 = (wave * 2 + i) * 16;
    f32x8 acc = {};
#pragma unroll
    for (int kk = 0; kk < 256; kk += 32) {
      FragBF a = load_a(sH32, lane & 15, 256, kk, lane);
      FragBF b = load_b(w2, n0 + (lane & 15), 256, kk, lane);
      acc = wmma_bf16(a, b, acc);
    }
    const int n = n0 + (lane & 15);
#pragma unroll
    for (int r = 0; r < 8; ++r) {
      const int m = r + ((lane & 16) ? 8 : 0);
      const float z = elu_f(acc[r] + b2[n]);
      outbf[(size_t)(row0 + m) * 256 + n] = f2bf_bits(z);
    }
  }
}

// ---------------------------------------------------------------- S5: GEMM 256x256 (LSTM input)
// A tile (16 rows x 256 bf16, contiguous 8KB) staged into LDS via the CDNA5
// Tensor Data Mover; D# built per CDNA5 ISA bit layout.
// 6-arg builtin form (clang-23 / therock-10.0): (g0, g1, g2, g3, g4, cpol).
__global__ __launch_bounds__(256) void k_gemm256(
    const uint32_t* __restrict__ Abf, const uint32_t* __restrict__ Wbf,
    uint16_t* __restrict__ outbf) {
  __shared__ __align__(16) __bf16 sA[16 * 256];
  const int row0 = blockIdx.x * 16;
  const int tid = threadIdx.x, lane = tid & 31, wave = tid >> 5;

#if __has_builtin(__builtin_amdgcn_tensor_load_to_lds)
  if (wave == 0) {
    const uint64_t gaddr = (uint64_t)(uintptr_t)Abf + (uint64_t)row0 * 512u;
    const uint32_t laddr = (uint32_t)(uintptr_t)(void*)&sA[0];
    u32x4 g0;
    g0.x = 1u;                                           // count=1 (valid user D#)
    g0.y = laddr;                                        // lds_addr [63:32]
    g0.z = (uint32_t)(gaddr & 0xFFFFFFFFu);              // global_addr [95:64]
    g0.w = (uint32_t)((gaddr >> 32) & 0x01FFFFFFu)       // global_addr[56:32]
           | (2u << 30);                                 // type=2 ("image") [127:126]
    i32x8 g1;
    g1[0] = (int)(1u << 16);                             // data_size=2B; mask=0
    g1[1] = (int)(256u << 16);                           // tensor_dim0[15:0] @ [63:48]
    g1[2] = (int)(16u << 16);                            // dim0 hi=0 | tensor_dim1=16
    g1[3] = (int)(256u << 16);                           // dim1 hi=0 | tile_dim0=256
    g1[4] = 16;                                          // tile_dim1=16, tile_dim2=0
    g1[5] = 256;                                         // tensor_dim0_stride lo
    g1[6] = 0;                                           // stride hi | dim1_stride lo
    g1[7] = 0;
    i32x4 gz4 = {0, 0, 0, 0};
    i32x8 gz8 = {0, 0, 0, 0, 0, 0, 0, 0};
    __builtin_amdgcn_tensor_load_to_lds(g0, g1, gz4, gz4, gz8, 0);
    __builtin_amdgcn_s_wait_tensorcnt(0);                // s_wait_tensorcnt 0
  }
  __syncthreads();
#else
  for (int i = tid; i < 16 * 128; i += 256)              // fallback: cooperative load
    ((uint32_t*)sA)[i] = Abf[(size_t)row0 * 128 + i];
  __syncthreads();
#endif

  const uint32_t* sA32 = (const uint32_t*)sA;
#pragma unroll
  for (int i = 0; i < 2; ++i) {
    const int n0 = (wave * 2 + i) * 16;
    f32x8 acc = {};
#pragma unroll
    for (int kk = 0; kk < 256; kk += 32) {
      FragBF a = load_a(sA32, lane & 15, 256, kk, lane);
      FragBF b = load_b(Wbf, n0 + (lane & 15), 256, kk, lane);
      acc = wmma_bf16(a, b, acc);
    }
    const int n = n0 + (lane & 15);
#pragma unroll
    for (int r = 0; r < 8; ++r) {
      const int m = r + ((lane & 16) ? 8 : 0);
      outbf[(size_t)(row0 + m) * 256 + n] = f2bf_bits(acc[r]);
    }
  }
}

// ---------------------------------------------------------------- S6: LSTM (one direction)
__global__ __launch_bounds__(256) void k_lstm(
    const uint16_t* __restrict__ xwbf,   // [NSEQ*T][256] bf16, flat row = n*T+t
    const uint32_t* __restrict__ whhbf,  // [256][64] bf16
    const float* __restrict__ bih, const float* __restrict__ bhh,
    uint16_t* __restrict__ outbf,        // [NSEQ*T][64] bf16
    int reverse) {
  __shared__ __bf16 sH[16 * 64];
  __shared__ float sC[16 * 64];
  __shared__ float sZ[16 * 256];
  __shared__ float sB[256];
  const int seq0 = blockIdx.x * 16;
  const int tid = threadIdx.x, lane = tid & 31, wave = tid >> 5;

  sB[tid] = bih[tid] + bhh[tid];
  for (int i = tid; i < 16 * 64; i += 256) {
    union { uint16_t u; __bf16 b; } c; c.u = 0;
    sH[i] = c.b;
    sC[i] = 0.f;
  }
  __syncthreads();

  const uint32_t* sH32 = (const uint32_t*)sH;
  for (int step = 0; step < TT; ++step) {
    const int t = reverse ? (TT - 1 - step) : step;
#pragma unroll
    for (int i = 0; i < 2; ++i) {                  // z = h @ whh.T  (K=64)
      const int n0 = (wave * 2 + i) * 16;
      f32x8 acc = {};
#pragma unroll
      for (int kk = 0; kk < 64; kk += 32) {
        FragBF a = load_a(sH32, lane & 15, 64, kk, lane);
        FragBF b = load_b(whhbf, n0 + (lane & 15), 64, kk, lane);
        acc = wmma_bf16(a, b, acc);
      }
      const int n = n0 + (lane & 15);
#pragma unroll
      for (int r = 0; r < 8; ++r) {
        const int m = r + ((lane & 16) ? 8 : 0);
        const size_t xrow = (size_t)(seq0 + m) * TT + t;
        sZ[m * 256 + n] = acc[r] + bf_bits2f(xwbf[xrow * 256 + n]) + sB[n];
      }
    }
    __syncthreads();
#pragma unroll
    for (int j = 0; j < 4; ++j) {                  // gates: 16 rows x 64 rh
      const int cell = tid + j * 256;
      const int m = cell >> 6, r = cell & 63;
      const float zi = sZ[m * 256 + r];
      const float zf = sZ[m * 256 + 64 + r];
      const float zg = sZ[m * 256 + 128 + r];
      const float zo = sZ[m * 256 + 192 + r];
      const float c = sigm_f(zf) * sC[cell] + sigm_f(zi) * tanhf(zg);
      const float h = sigm_f(zo) * tanhf(c);
      sC[cell] = c;
      union { uint16_t u; __bf16 b; } cv; cv.u = f2bf_bits(h);
      sH[cell] = cv.b;
      outbf[((size_t)(seq0 + m) * TT + t) * 64 + r] = cv.u;
    }
    __syncthreads();
  }
}

// ---------------------------------------------------------------- S7: output heads
__global__ __launch_bounds__(256) void k_head(
    const uint16_t* __restrict__ fwdbf, const uint16_t* __restrict__ revbf,
    const uint32_t* __restrict__ ew1bf, const float* __restrict__ eb1,
    const float* __restrict__ ew2, const float* __restrict__ eb2,
    const float* __restrict__ pw, const float* __restrict__ pb,
    float* __restrict__ out) {
  __shared__ __bf16 sA[16 * 128];
  __shared__ float sHid[16 * 256];
  const int row0 = blockIdx.x * 16;
  const int tid = threadIdx.x, lane = tid & 31, wave = tid >> 5;

  uint32_t* sA32w = (uint32_t*)sA;
  const uint32_t* fwd32 = (const uint32_t*)fwdbf;
  const uint32_t* rev32 = (const uint32_t*)revbf;
  for (int i = tid; i < 16 * 64; i += 256) {       // 16 rows x 64 dwords (128 bf16)
    const int m = i >> 6, kd = i & 63;
    uint32_t val;
    if (kd < 32) val = fwd32[(size_t)(row0 + m) * 32 + kd];
    else         val = rev32[(size_t)(row0 + m) * 32 + (kd - 32)];
    sA32w[m * 64 + kd] = val;
  }
  __syncthreads();

  const uint32_t* sA32 = (const uint32_t*)sA;
#pragma unroll
  for (int i = 0; i < 2; ++i) {                    // encoder layer 1, K=128
    const int n0 = (wave * 2 + i) * 16;
    f32x8 acc = {};
#pragma unroll
    for (int kk = 0; kk < 128; kk += 32) {
      FragBF a = load_a(sA32, lane & 15, 128, kk, lane);
      FragBF b = load_b(ew1bf, n0 + (lane & 15), 128, kk, lane);
      acc = wmma_bf16(a, b, acc);
    }
    const int n = n0 + (lane & 15);
#pragma unroll
    for (int r = 0; r < 8; ++r) {
      const int m = r + ((lane & 16) ? 8 : 0);
      sHid[m * 256 + n] = elu_f(acc[r] + eb1[n]);
    }
  }
  __syncthreads();

  if (tid < 32) {                                  // enc: 16 rows x 2 outputs
    const int m = tid >> 1, ko = tid & 1;
    float s = eb2[ko];
    for (int j = 0; j < 256; ++j) s += sHid[m * 256 + j] * ew2[ko * 256 + j];
    const int flat = row0 + m;
    const int t = flat & (TT - 1);
    const int be = flat >> 6;
    const int e = be % EE;
    const int b = be / EE;
    out[OUT_HALF + ((((size_t)b * TT + t) * EE + e) * KK + ko)] = s;
  } else if (tid < 64) {                           // prior: fwd @ p_w.T + p_b
    const int q = tid - 32;
    const int m = q >> 1, ko = q & 1;
    float s = pb[ko];
    for (int j = 0; j < 64; ++j)
      s += bf_bits2f(fwdbf[(size_t)(row0 + m) * 64 + j]) * pw[ko * 64 + j];
    const int flat = row0 + m;
    const int t = flat & (TT - 1);
    const int be = flat >> 6;
    const int e = be % EE;
    const int b = be / EE;
    out[(((size_t)b * TT + t) * EE + e) * KK + ko] = s;
  }
}

// ---------------------------------------------------------------- launch
extern "C" void kernel_launch(void* const* d_in, const int* in_sizes, int n_in,
                              void* d_out, int out_size, void* d_ws, size_t ws_size,
                              hipStream_t stream) {
  const float* rel_feat = (const float*)d_in[0];
  const float* edge_attr_raw = (const float*)d_in[1];
  const float* edge_pos = (const float*)d_in[2];
  const float* ef_w1 = (const float*)d_in[3];
  const float* ef_b1 = (const float*)d_in[4];
  const float* ef_w2 = (const float*)d_in[5];
  const float* ef_b2 = (const float*)d_in[6];
  const float* res_w = (const float*)d_in[7];
  const float* res_b = (const float*)d_in[8];
  const float* m3_w1 = (const float*)d_in[9];
  const float* m3_b1 = (const float*)d_in[10];
  const float* m3_w2 = (const float*)d_in[11];
  const float* m3_b2 = (const float*)d_in[12];
  const float* m4_w1 = (const float*)d_in[13];
  const float* m4_b1 = (const float*)d_in[14];
  const float* m4_w2 = (const float*)d_in[15];
  const float* m4_b2 = (const float*)d_in[16];
  const float* f_wih = (const float*)d_in[17];
  const float* f_whh = (const float*)d_in[18];
  const float* f_bih = (const float*)d_in[19];
  const float* f_bhh = (const float*)d_in[20];
  const float* r_wih = (const float*)d_in[21];
  const float* r_whh = (const float*)d_in[22];
  const float* r_bih = (const float*)d_in[23];
  const float* r_bhh = (const float*)d_in[24];
  const float* p_w = (const float*)d_in[25];
  const float* p_b = (const float*)d_in[26];
  const float* e_w1 = (const float*)d_in[27];
  const float* e_b1 = (const float*)d_in[28];
  const float* e_w2 = (const float*)d_in[29];
  const float* e_b2 = (const float*)d_in[30];
  (void)in_sizes; (void)n_in; (void)out_size; (void)ws_size;

  uint16_t* W = (uint16_t*)d_ws;
  auto u32 = [&](size_t off) { return (const uint32_t*)(W + off); };
  auto cvt = [&](const float* src, size_t off, int n) {
    k_f2bf<<<(n + 255) / 256, 256, 0, stream>>>(src, W + off, n);
  };

  // S0: weight conversion to bf16
  k_pad_efw1<<<(256 * 32 + 255) / 256, 256, 0, stream>>>(ef_w1, W + O_EFW1);
  cvt(ef_w2, O_EFW2, 256 * 256);
  cvt(m3_w1, O_M3W1, 256 * 256);
  cvt(m3_w2, O_M3W2, 256 * 256);
  cvt(m4_w1, O_M4W1, 256 * 768);
  cvt(m4_w2, O_M4W2, 256 * 256);
  cvt(f_wih, O_FWIH, 256 * 256);
  cvt(r_wih, O_RWIH, 256 * 256);
  cvt(f_whh, O_FWHH, 256 * 64);
  cvt(r_whh, O_RWHH, 256 * 64);
  cvt(e_w1,  O_EW1,  256 * 128);

  // S1: edge filter MLP
  k_edge_mlp<<<EROWS / 16, 256, 0, stream>>>(
      edge_attr_raw, edge_pos, u32(O_EFW1), ef_b1, u32(O_EFW2), ef_b2, W + O_EDGE);
  // S2: edge->node aggregation + residual
  k_aggregate<<<NROWS, 256, 0, stream>>>(W + O_EDGE, rel_feat, res_w, res_b, W + O_NIN);
  // S3: node MLP
  k_mlp2_256<<<NROWS / 16, 256, 0, stream>>>(
      u32(O_NIN), u32(O_M3W1), m3_b1, u32(O_M3W2), m3_b2, W + O_NODE);
  // S4: node2edge concat MLP
  k_mlp4<<<EROWS / 16, 256, 0, stream>>>(
      u32(O_NODE), u32(O_EDGE), u32(O_M4W1), m4_b1, u32(O_M4W2), m4_b2, W + O_XE);
  // S5: LSTM input GEMMs (TDM-staged A tiles)
  k_gemm256<<<EROWS / 16, 256, 0, stream>>>(u32(O_XE), u32(O_FWIH), W + O_XWF);
  k_gemm256<<<EROWS / 16, 256, 0, stream>>>(u32(O_XE), u32(O_RWIH), W + O_XWR);
  // S6: bidirectional LSTM
  k_lstm<<<NSEQ / 16, 256, 0, stream>>>(W + O_XWF, u32(O_FWHH), f_bih, f_bhh, W + O_FWD, 0);
  k_lstm<<<NSEQ / 16, 256, 0, stream>>>(W + O_XWR, u32(O_RWHH), r_bih, r_bhh, W + O_REV, 1);
  // S7: heads -> prior / enc
  k_head<<<EROWS / 16, 256, 0, stream>>>(
      W + O_FWD, W + O_REV, u32(O_EW1), e_b1, e_w2, e_b2, p_w, p_b, (float*)d_out);
}
